// MambaBlock_12369505812723
// MI455X (gfx1250) — matrix operand order repaired and checked by reading, not compiled, not run on hardware
//
#include <hip/hip_runtime.h>

// ---------------------------------------------------------------------------
// Mamba block for MI455X (gfx1250, wave32, WMMA + Tensor Data Mover).
// All GEMM operands are stored bf16 in global memory (converted once by the
// producers), so GEMM tile staging is a pure copy: TDM tensor_load_to_lds
// double-buffered per K-step (fallback: b128 loads) and 4 v_wmma per wave
// per K-step on a 128x64x32 block tile.
// ---------------------------------------------------------------------------

#ifndef __has_builtin
#define __has_builtin(x) 0
#endif
#if __has_builtin(__builtin_amdgcn_tensor_load_to_lds) && \
    __has_builtin(__builtin_amdgcn_s_wait_tensorcnt)
#define HAVE_TDM 1
#else
#define HAVE_TDM 0
#endif

typedef __attribute__((ext_vector_type(16))) __bf16    v16bf;
typedef __attribute__((ext_vector_type(8)))  float     v8f;
typedef __attribute__((ext_vector_type(4)))  unsigned  v4u;
typedef __attribute__((ext_vector_type(8)))  int       v8i;
typedef __attribute__((ext_vector_type(4)))  int       v4i;

union Frag { v16bf v; unsigned long long u[4]; };

__device__ __forceinline__ unsigned f2bf(float f) {
  unsigned u = __float_as_uint(f);
  u += 0x7FFFu + ((u >> 16) & 1u);  // round-to-nearest-even
  return u >> 16;
}
__device__ __forceinline__ unsigned pk2bf(float a, float b) {
  return f2bf(a) | (f2bf(b) << 16);
}

#if HAVE_TDM
// Issue one 2-D bf16 tile DMA: global [dim1 x dim0] (stride elems) -> LDS.
// D# layout per CDNA5 ISA 8.3/8.4: group0 = {count, lds_addr, global_addr,
// type=2}; group1 = {data_size=2B, tensor_dim0/1, tile_dim0/1, dim0_stride}.
// OOB (dim0 < tile_w or dim1 < tile_h) reads return zero -> free padding.
__device__ __forceinline__ void tdm_load_2d_bf16(
    unsigned lds_off, const unsigned short* gptr,
    unsigned dim0, unsigned dim1, unsigned stride,
    unsigned tile_w, unsigned tile_h)
{
  unsigned long long ga = (unsigned long long)gptr;
  v4u g0;
  g0[0] = 1u;                                 // count=1 (valid descriptor)
  g0[1] = lds_off;                            // LDS byte address
  g0[2] = (unsigned)ga;                       // global_addr[31:0]
  g0[3] = (unsigned)(ga >> 32) | (2u << 30);  // global_addr[56:32] | type=2
  v8i g1;
  g1[0] = (int)(1u << 16);                    // data_size=1 -> 2 bytes
  g1[1] = (int)(dim0 << 16);                  // tensor_dim0[15:0]
  g1[2] = (int)((dim0 >> 16) | (dim1 << 16)); // tensor_dim0[31:16]|dim1[15:0]
  g1[3] = (int)((dim1 >> 16) | (tile_w << 16)); // dim1[31:16] | tile_dim0
  g1[4] = (int)tile_h;                        // tile_dim1 (tile_dim2 = 0)
  g1[5] = (int)stride;                        // tensor_dim0_stride[31:0]
  g1[6] = 0;
  g1[7] = 0;
  v4i zz = {0, 0, 0, 0};
#if defined(__clang_major__) && (__clang_major__ >= 23)
  v8i z8 = {};
  __builtin_amdgcn_tensor_load_to_lds(g0, g1, zz, zz, z8, 0);
#else
  __builtin_amdgcn_tensor_load_to_lds(g0, g1, zz, zz, 0);
#endif
}
#endif

// WMODE: 0 = f32 out only, 1 = f32 + bf16 out, 2 = bf16 out only.
// EPI:   0 = none, 2 = exact GELU, 3 = softplus.
template <int EPI, int N, int K, int WMODE>
__global__ __launch_bounds__(256) void gemm_bf16_wmma(
    const unsigned short* __restrict__ A, int lda,  // bf16 [M, lda]
    const unsigned short* __restrict__ W,           // bf16 [N, K]
    const float* __restrict__ bias,                 // [N] or nullptr
    float* __restrict__ Cf,                         // [M, N] (WMODE != 2)
    unsigned short* __restrict__ Cb)                // bf16 [M, N] (WMODE != 0)
{
  constexpr int BM = 128, BN = 64, BK = 32;
  constexpr int KT = (K + BK - 1) / BK;
#if HAVE_TDM
  __shared__ __align__(16) unsigned short As[2][BM * BK];
  __shared__ __align__(16) unsigned short Bs[2][BN * BK];
#else
  __shared__ __align__(16) unsigned short As[1][BM * BK];
  __shared__ __align__(16) unsigned short Bs[1][BN * BK];
#endif

  const int tid  = threadIdx.x;
  const int lane = tid & 31;
  const int wave = tid >> 5;     // 8 waves
  const int wm   = wave >> 1;    // 0..3 -> 32-row stripe
  const int wn   = wave & 1;     // 0..1 -> 32-col stripe
  const int m0   = blockIdx.y * BM;
  const int n0   = blockIdx.x * BN;

  v8f acc[2][2] = {{{}, {}}, {{}, {}}};

#if HAVE_TDM
  auto issue = [&](int kt, int buf) {
    const int k0 = kt * BK;
    tdm_load_2d_bf16((unsigned)(unsigned long long)&As[buf][0],
                     A + (long)m0 * lda + k0,
                     (unsigned)(K - k0), (unsigned)BM, (unsigned)lda, BK, BM);
    tdm_load_2d_bf16((unsigned)(unsigned long long)&Bs[buf][0],
                     W + (long)n0 * K + k0,
                     (unsigned)(K - k0), (unsigned)(N - n0), (unsigned)K, BK, BN);
  };
  if (wave == 0) issue(0, 0);
#else
  const int arow = tid >> 1, acol = (tid & 1) * 16;
  const int brow = tid >> 2, bcol = (tid & 3) * 8;
  const bool nok = (N % BN == 0) || (n0 + brow < N);
  const unsigned short* Aptr = A + (long)(m0 + arow) * lda + acol;
  const unsigned short* Wptr = W + (long)(n0 + brow) * K + bcol;
#endif

  for (int kt = 0; kt < KT; ++kt) {
#if HAVE_TDM
    const int cur = kt & 1;
    if (wave == 0) {
      if (kt + 1 < KT) {
        issue(kt + 1, cur ^ 1);                    // prefetch next tile
        __builtin_amdgcn_s_wait_tensorcnt(2);      // current tile's 2 DMAs done
      } else {
        __builtin_amdgcn_s_wait_tensorcnt(0);
      }
    }
    __syncthreads();   // release staged buffer to all waves
#else
    const int cur = 0;
    const int k0 = kt * BK;
    {
      unsigned long long va[4] = {0, 0, 0, 0};
      #pragma unroll
      for (int c = 0; c < 2; ++c) {
        if ((K % BK == 0) || (k0 + acol + c * 8 < K)) {  // folds if K%32==0
          const unsigned long long* p =
              (const unsigned long long*)(Aptr + k0 + c * 8);
          va[c * 2] = p[0]; va[c * 2 + 1] = p[1];
        }
      }
      unsigned long long* da = (unsigned long long*)(&As[0][arow * BK + acol]);
      da[0] = va[0]; da[1] = va[1]; da[2] = va[2]; da[3] = va[3];
      unsigned long long vb[2] = {0, 0};
      if (nok && ((K % BK == 0) || (k0 + bcol < K))) {
        const unsigned long long* p = (const unsigned long long*)(Wptr + k0);
        vb[0] = p[0]; vb[1] = p[1];
      }
      unsigned long long* db = (unsigned long long*)(&Bs[0][brow * BK + bcol]);
      db[0] = vb[0]; db[1] = vb[1];
    }
    __syncthreads();
#endif
    const unsigned short* curA = As[cur];
    const unsigned short* curB = Bs[cur];

    // A frag (16-bit A 16x32): lane<16 -> K {0..7,16..23}; lane>=16 -> +8.
    const int kbA = (lane >= 16) ? 8 : 0;
    Frag fa[2];
    #pragma unroll
    for (int ms = 0; ms < 2; ++ms) {
      const int fm = wm * 32 + ms * 16 + (lane & 15);
      const unsigned long long* p0 =
          (const unsigned long long*)(curA + fm * BK + kbA);
      const unsigned long long* p1 =
          (const unsigned long long*)(curA + fm * BK + kbA + 16);
      fa[ms].u[0] = p0[0]; fa[ms].u[1] = p0[1];
      fa[ms].u[2] = p1[0]; fa[ms].u[3] = p1[1];
    }
    // B frag (32x16): lane<16 -> K {0..7,16..23}; lane>=16 -> +8.
    const int kbB = (lane >= 16) ? 8 : 0;
    Frag fb[2];
    #pragma unroll
    for (int ns = 0; ns < 2; ++ns) {
      const int fn = wn * 32 + ns * 16 + (lane & 15);
      const unsigned long long* p =
          (const unsigned long long*)(curB + fn * BK + kbB);
      const unsigned long long* p1 =
          (const unsigned long long*)(curB + fn * BK + kbB + 16);
      fb[ns].u[0] = p[0]; fb[ns].u[1] = p[1];
      fb[ns].u[2] = p1[0]; fb[ns].u[3] = p1[1];
    }
    #pragma unroll
    for (int ms = 0; ms < 2; ++ms)
      #pragma unroll
      for (int ns = 0; ns < 2; ++ns)
        acc[ms][ns] = __builtin_amdgcn_wmma_f32_16x16x32_bf16(
            false, fa[ms].v, false, fb[ns].v, (short)0, acc[ms][ns],
            false, false);
    __syncthreads();
  }

  // Epilogue. C/D layout: VGPR v, lane l -> m = v + 8*(l>>4), n = (l&15).
  #pragma unroll
  for (int ms = 0; ms < 2; ++ms) {
    #pragma unroll
    for (int ns = 0; ns < 2; ++ns) {
      const int gn = n0 + wn * 32 + ns * 16 + (lane & 15);
      if ((N % BN == 0) || (gn < N)) {
        const float bv = bias ? bias[gn] : 0.0f;
        #pragma unroll
        for (int v = 0; v < 8; ++v) {
          const int gm = m0 + wm * 32 + ms * 16 + v + 8 * (lane >> 4);
          float c = acc[ms][ns][v] + bv;
          if (EPI == 2) c = 0.5f * c * (1.0f + erff(c * 0.70710678118654752f));
          else if (EPI == 3) c = (c > 20.0f) ? c : log1pf(__expf(c));
          if (WMODE != 2) Cf[(long)gm * N + gn] = c;
          if (WMODE != 0) Cb[(long)gm * N + gn] = (unsigned short)f2bf(c);
        }
      }
    }
  }
}

// Bulk fp32 -> bf16 (8 elems/thread, element count multiple of 2048).
__global__ __launch_bounds__(256) void f32_to_bf16_vec8(
    const float* __restrict__ in, unsigned short* __restrict__ out)
{
  const long i = ((long)blockIdx.x * 256 + threadIdx.x) * 8;
  const float4 a = *(const float4*)(in + i);
  const float4 b = *(const float4*)(in + i + 4);
  uint4 o;
  o.x = pk2bf(a.x, a.y); o.y = pk2bf(a.z, a.w);
  o.z = pk2bf(b.x, b.y); o.w = pk2bf(b.z, b.w);
  *(uint4*)(out + i) = o;
}

// Depthwise causal conv (DC=4) + bias + SiLU; writes f32 (scan) + bf16 (x_proj A).
__global__ __launch_bounds__(256) void conv_silu_kernel(
    const float* __restrict__ xz, const float* __restrict__ conv_w,
    const float* __restrict__ conv_b, float* __restrict__ xic,
    unsigned short* __restrict__ xicbf)
{
  const int idx = blockIdx.x * 256 + threadIdx.x;  // B*L*512
  const int d   = idx & 511;
  const int row = idx >> 9;     // b*L + l
  const int l   = row & 2047;
  const int bL  = row - l;
  float s = conv_b[d];
  #pragma unroll
  for (int j = 0; j < 4; ++j) {
    const int li = l - 3 + j;
    if (li >= 0) s += conv_w[d * 4 + j] * xz[(long)(bL + li) * 1024 + d];
  }
  const float r = s / (1.0f + __expf(-s));
  xic[(long)row * 512 + d]   = r;
  xicbf[(long)row * 512 + d] = (unsigned short)f2bf(r);
}

// Selective scan. One lane per (b,d,n); 16-lane groups reduce over n.
// Output (gated y) is consumed only by out_proj -> bf16 only.
__global__ __launch_bounds__(256) void ssm_scan_kernel(
    const float* __restrict__ dtb,    // [B*L, 512] softplus(dt)
    const float* __restrict__ xic,    // [B*L, 512]
    const float* __restrict__ xdbl,   // [B*L, 48]: [dt16 | B16 | C16]
    const float* __restrict__ xz,     // [B*L, 1024]: z at col 512+d
    const float* __restrict__ A_log,  // [512, 16]
    const float* __restrict__ D_ssm,  // [512]
    unsigned short* __restrict__ ygbf)// bf16 [B*L, 512]
{
  const int t = blockIdx.x * 256 + threadIdx.x;   // 32768 lanes
  const int n = t & 15;
  const int p = t >> 4;       // b*512 + d
  const int d = p & 511;
  const int b = p >> 9;
  const float Aval = -__expf(A_log[d * 16 + n]);
  const float Dval = D_ssm[d];
  float h = 0.0f;
  const long base = (long)b * 2048;
  for (int l = 0; l < 2048; ++l) {
    const long r = base + l;
    if (l + 16 < 2048) {  // uniform branch: prefetch the sequential streams
      __builtin_prefetch(dtb + (r + 16) * 512 + d, 0, 1);
      __builtin_prefetch(xic + (r + 16) * 512 + d, 0, 1);
    }
    const float dt = dtb[r * 512 + d];
    const float xi = xic[r * 512 + d];
    const float Bn = xdbl[r * 48 + 16 + n];
    const float Cn = xdbl[r * 48 + 32 + n];
    h = __expf(dt * Aval) * h + dt * Bn * xi;
    float val = h * Cn;
    val += __shfl_xor(val, 1, 32);
    val += __shfl_xor(val, 2, 32);
    val += __shfl_xor(val, 4, 32);
    val += __shfl_xor(val, 8, 32);
    if (n == 0) {
      const float z = xz[r * 1024 + 512 + d];
      const float y = val + xi * Dval;
      ygbf[r * 512 + d] =
          (unsigned short)f2bf(y * (z / (1.0f + __expf(-z))));
    }
  }
}

// out = LayerNorm(a + b) over D=256; optional bf16 copy of the output.
__global__ __launch_bounds__(256) void ln_add_kernel(
    const float* __restrict__ a, const float* __restrict__ b,
    const float* __restrict__ g, const float* __restrict__ beta,
    float* __restrict__ out, unsigned short* __restrict__ outb)
{
  const int lane = threadIdx.x & 31;
  const int row  = blockIdx.x * 8 + (threadIdx.x >> 5);
  const float* pa = a + (long)row * 256;
  const float* pb = b + (long)row * 256;
  float r[8];
  float s = 0.0f;
  #pragma unroll
  for (int j = 0; j < 8; ++j) {
    const int c = lane + 32 * j;
    r[j] = pa[c] + pb[c];
    s += r[j];
  }
  #pragma unroll
  for (int o = 16; o > 0; o >>= 1) s += __shfl_xor(s, o, 32);
  const float mean = s * (1.0f / 256.0f);
  float v = 0.0f;
  #pragma unroll
  for (int j = 0; j < 8; ++j) { const float dlt = r[j] - mean; v += dlt * dlt; }
  #pragma unroll
  for (int o = 16; o > 0; o >>= 1) v += __shfl_xor(v, o, 32);
  const float inv = rsqrtf(v * (1.0f / 256.0f) + 1e-5f);
  float* po = out + (long)row * 256;
  #pragma unroll
  for (int j = 0; j < 8; ++j) {
    const int c = lane + 32 * j;
    const float o = (r[j] - mean) * inv * g[c] + beta[c];
    po[c] = o;
    if (outb) outb[(long)row * 256 + c] = (unsigned short)f2bf(o);
  }
}

extern "C" void kernel_launch(void* const* d_in, const int* in_sizes, int n_in,
                              void* d_out, int out_size, void* d_ws, size_t ws_size,
                              hipStream_t stream)
{
  const float* x      = (const float*)d_in[0];
  const float* in_w   = (const float*)d_in[1];
  const float* in_b   = (const float*)d_in[2];
  const float* conv_w = (const float*)d_in[3];
  const float* conv_b = (const float*)d_in[4];
  const float* xp_w   = (const float*)d_in[5];
  const float* dt_w   = (const float*)d_in[6];
  const float* dt_b   = (const float*)d_in[7];
  const float* A_log  = (const float*)d_in[8];
  const float* D_ssm  = (const float*)d_in[9];
  const float* out_w  = (const float*)d_in[10];
  const float* out_b  = (const float*)d_in[11];
  const float* ln1_g  = (const float*)d_in[12];
  const float* ln1_b  = (const float*)d_in[13];
  const float* fc1_w  = (const float*)d_in[14];
  const float* fc1_b  = (const float*)d_in[15];
  const float* fc2_w  = (const float*)d_in[16];
  const float* fc2_b  = (const float*)d_in[17];
  const float* ln2_g  = (const float*)d_in[18];
  const float* ln2_b  = (const float*)d_in[19];
  float* out = (float*)d_out;

  const int M = 8192;  // B*L
  // ---- f32 workspace ----
  float* ws   = (float*)d_ws;
  float* xz   = ws;                       // [8192,1024]
  float* xic  = xz   + (long)M * 1024;    // [8192,512]
  float* xdbl = xic  + (long)M * 512;     // [8192,48]
  float* dtb  = xdbl + (long)M * 48;      // [8192,512]
  float* ssm  = dtb  + (long)M * 512;     // [8192,256]
  float* y1   = ssm  + (long)M * 256;     // [8192,256]
  float* ffn  = ssm;                      // reuse: ssm dead after LN1
  // ---- bf16 workspace ----
  unsigned short* u = (unsigned short*)(y1 + (long)M * 256);
  unsigned short* xbf    = u; u += (long)M * 256;
  unsigned short* xicbf  = u; u += (long)M * 512;
  unsigned short* xdblbf = u; u += (long)M * 48;
  unsigned short* ygbf   = u; u += (long)M * 512;
  unsigned short* w_in   = u; u += 1024 * 256;
  unsigned short* w_xp   = u; u += 48 * 512;
  unsigned short* w_dt   = u; u += 512 * 16;
  unsigned short* w_out  = u; u += 256 * 512;
  unsigned short* w_f1   = u; u += 1024 * 256;
  unsigned short* w_f2   = u; u += 256 * 1024;
  unsigned short* h1bf   = (unsigned short*)xz;  // reuse: xz dead after scan
  unsigned short* y1bf   = ygbf;                 // reuse: ygbf dead after out_proj
  // total: ~85 MB f32 + ~24 MB bf16

  const dim3 blk(256);
  // 0. one-time fp32 -> bf16 conversions (x + all six weights)
  f32_to_bf16_vec8<<<dim3(M * 256 / 2048), blk, 0, stream>>>(x, xbf);
  f32_to_bf16_vec8<<<dim3(1024 * 256 / 2048), blk, 0, stream>>>(in_w, w_in);
  f32_to_bf16_vec8<<<dim3(48 * 512 / 2048), blk, 0, stream>>>(xp_w, w_xp);
  f32_to_bf16_vec8<<<dim3(512 * 16 / 2048), blk, 0, stream>>>(dt_w, w_dt);
  f32_to_bf16_vec8<<<dim3(256 * 512 / 2048), blk, 0, stream>>>(out_w, w_out);
  f32_to_bf16_vec8<<<dim3(1024 * 256 / 2048), blk, 0, stream>>>(fc1_w, w_f1);
  f32_to_bf16_vec8<<<dim3(256 * 1024 / 2048), blk, 0, stream>>>(fc2_w, w_f2);

  // 1. in_proj -> xz (f32; consumed by conv + z-gate)
  gemm_bf16_wmma<0, 1024, 256, 0><<<dim3(16, M / 128), blk, 0, stream>>>(
      xbf, 256, w_in, in_b, xz, nullptr);
  // 2. conv + SiLU -> xic (f32 for scan) + xicbf (A of x_proj)
  conv_silu_kernel<<<dim3((M * 512) / 256), blk, 0, stream>>>(
      xz, conv_w, conv_b, xic, xicbf);
  // 3. x_proj -> x_dbl (f32 for scan B/C) + bf16 (A of dt_proj)
  gemm_bf16_wmma<0, 48, 512, 1><<<dim3(1, M / 128), blk, 0, stream>>>(
      xicbf, 512, w_xp, nullptr, xdbl, xdblbf);
  // 4. dt_proj + softplus -> dtb (f32; scan only)
  gemm_bf16_wmma<3, 512, 16, 0><<<dim3(8, M / 128), blk, 0, stream>>>(
      xdblbf, 48, w_dt, dt_b, dtb, nullptr);
  // 5. selective scan + D-skip + SiLU(z) gate -> ygbf (bf16; out_proj A)
  ssm_scan_kernel<<<dim3(32768 / 256), blk, 0, stream>>>(
      dtb, xic, xdbl, xz, A_log, D_ssm, ygbf);
  // 6. out_proj -> ssm (f32)
  gemm_bf16_wmma<0, 256, 512, 0><<<dim3(4, M / 128), blk, 0, stream>>>(
      ygbf, 512, w_out, out_b, ssm, nullptr);
  // 7. LN1(x + ssm) -> y1 (f32 residual) + y1bf (fc1 A)
  ln_add_kernel<<<dim3(M / 8), blk, 0, stream>>>(x, ssm, ln1_g, ln1_b, y1, y1bf);
  // 8. fc1 + GELU -> h1bf (bf16 only; consumed only by fc2)
  gemm_bf16_wmma<2, 1024, 256, 2><<<dim3(16, M / 128), blk, 0, stream>>>(
      y1bf, 256, w_f1, fc1_b, nullptr, h1bf);
  // 9. fc2 + GELU -> ffn (f32)
  gemm_bf16_wmma<2, 256, 1024, 0><<<dim3(4, M / 128), blk, 0, stream>>>(
      h1bf, 1024, w_f2, fc2_b, ffn, nullptr);
  // 10. LN2(y1 + ffn) -> out
  ln_add_kernel<<<dim3(M / 8), blk, 0, stream>>>(y1, ffn, ln2_g, ln2_b, out, nullptr);
}
